// GraphNet_49847390437921
// MI455X (gfx1250) — matrix operand order repaired and checked by reading, not compiled, not run on hardware
//
#include <hip/hip_runtime.h>
#include <hip/hip_bf16.h>

// ---------------------------------------------------------------------------
// MEGNet-style graph network for MI455X (gfx1250, wave32, WMMA).
// Heavy GEMMs (rows x 64) @ (64x64) run as v_wmma_f32_16x16x32_bf16 with f32
// accumulation; h_e0/h_u terms are algebraically folded into per-row scalar *
// precomputed 64-vectors to kill an entire GEMM + 205MB/round stream.
// h_e lives in d_out (in-place, row-local). h_v (25.6MB) is L2-resident for
// the random bond gathers.
// ---------------------------------------------------------------------------

typedef __attribute__((ext_vector_type(16))) __bf16 v16bf;
typedef __attribute__((ext_vector_type(8)))  float  v8f;

#define D_FEAT 64
#define TILE_M 16
#define WAVES_PER_BLOCK 8
#define NMATS 3                       // 3 GEMM sources per update
#define FRAG_ELEMS 512                // 32 lanes * 16 bf16 per B fragment
#define WFRAG_TOTAL (NMATS * 2 * 4 * FRAG_ELEMS)   // [mat][kt][nt][lane][i]

// ---- A fragment: 16-bit A 16x32 layout (ISA 7.12.2) -----------------------
// lanes 0-15 : M=lane,   elems 0..7 -> K=base+0..7,  elems 8..15 -> K=base+16..23
// lanes 16-31: M=lane-16, same +8 offset.  fb = kt*32 + halfsel*8.
__device__ inline v16bf make_a_frag(const float* __restrict__ rowp, int fb) {
  float4 a0 = *(const float4*)(rowp + fb);
  float4 a1 = *(const float4*)(rowp + fb + 4);
  float4 a2 = *(const float4*)(rowp + fb + 16);
  float4 a3 = *(const float4*)(rowp + fb + 20);
  v16bf A;
  A[0]  = (__bf16)a0.x; A[1]  = (__bf16)a0.y; A[2]  = (__bf16)a0.z; A[3]  = (__bf16)a0.w;
  A[4]  = (__bf16)a1.x; A[5]  = (__bf16)a1.y; A[6]  = (__bf16)a1.z; A[7]  = (__bf16)a1.w;
  A[8]  = (__bf16)a2.x; A[9]  = (__bf16)a2.y; A[10] = (__bf16)a2.z; A[11] = (__bf16)a2.w;
  A[12] = (__bf16)a3.x; A[13] = (__bf16)a3.y; A[14] = (__bf16)a3.z; A[15] = (__bf16)a3.w;
  return A;
}

// One 64-K GEMM source accumulated into 4 16x16 f32 tiles (N = 64).
__device__ inline void wmma_source_accum(const __bf16* __restrict__ sW, int matsel,
                                         const float* __restrict__ rowp,
                                         int halfsel, int lane, v8f* acc) {
#pragma unroll
  for (int kt = 0; kt < 2; kt++) {
    v16bf A = make_a_frag(rowp, kt * 32 + halfsel * 8);
    const __bf16* fbase = sW + (size_t)((matsel * 2 + kt) * 4) * FRAG_ELEMS + lane * 16;
#pragma unroll
    for (int nt = 0; nt < 4; nt++) {
      v16bf B = *(const v16bf*)(fbase + nt * FRAG_ELEMS);
      acc[nt] = __builtin_amdgcn_wmma_f32_16x16x32_bf16(
          false, A, false, B, (short)0, acc[nt], false, false);
    }
  }
}

// ---------------------------------------------------------------------------
// Edge kernel: h_e_new = relu( [h_e | h_v[src] | h_v[dst]] @ Wcat
//                              + bo[b]*cvec + gvec )
// plus scatter-add into e_bar_i[src], e_bar_i[dst] and e_bar accumulation.
// ---------------------------------------------------------------------------
__global__ void __launch_bounds__(256) edge_kernel(
    const float* __restrict__ h_e_in, float* __restrict__ h_e_out,
    const float* __restrict__ h_v,
    const int* __restrict__ bond_idxs, const float* __restrict__ bond_orders,
    const __bf16* __restrict__ wfrag,
    const float* __restrict__ cvec, const float* __restrict__ gvec,
    float* __restrict__ e_bar_i, float* __restrict__ e_bar,
    int n_bonds, int use_he, int do_scatter) {
  __shared__ __align__(16) __bf16 sW[WFRAG_TOTAL];
  __shared__ float sEbar[D_FEAT];

  int tid = threadIdx.x;
  {  // stage pre-swizzled bf16 weights into LDS (24KB)
    const uint4* src = (const uint4*)wfrag;
    uint4* dst = (uint4*)sW;
    for (int i = tid; i < (WFRAG_TOTAL * 2) / 16; i += 256) dst[i] = src[i];
  }
  if (tid < D_FEAT) sEbar[tid] = 0.f;
  __syncthreads();

  int lane = tid & 31;
  int wave = tid >> 5;
  int halfsel = lane >> 4;
  int lm = lane & 15;
  int ntiles = n_bonds / TILE_M;
  int wstride = gridDim.x * WAVES_PER_BLOCK;

  float cvals[4], gvals[4], ebar_acc[4] = {0.f, 0.f, 0.f, 0.f};
#pragma unroll
  for (int nt = 0; nt < 4; nt++) {
    cvals[nt] = cvec[nt * 16 + lm];
    gvals[nt] = gvec[nt * 16 + lm];
  }

  for (int tile = blockIdx.x * WAVES_PER_BLOCK + wave; tile < ntiles; tile += wstride) {
    int bond0 = tile * TILE_M;
    int myrow = bond0 + lm;  // A-matrix row handled by this lane
    int2 myij = ((const int2*)bond_idxs)[myrow];

    v8f acc[4] = {};
    if (use_he)
      wmma_source_accum(sW, 0, h_e_in + (size_t)myrow * D_FEAT, halfsel, lane, acc);
    wmma_source_accum(sW, 1, h_v + (size_t)myij.x * D_FEAT, halfsel, lane, acc);
    wmma_source_accum(sW, 2, h_v + (size_t)myij.y * D_FEAT, halfsel, lane, acc);

    // Epilogue: C/D layout => lane holds (m = r + 8*halfsel, n = nt*16+lm)
    int rbase = bond0 + halfsel * 8;
#pragma unroll
    for (int r = 0; r < 8; r++) {
      int b = rbase + r;
      float bo = bond_orders[b];
      int2 ii = ((const int2*)bond_idxs)[b];
      size_t ob = (size_t)b * D_FEAT + lm;
#pragma unroll
      for (int nt = 0; nt < 4; nt++) {
        float v = acc[nt][r] + bo * cvals[nt] + gvals[nt];
        v = fmaxf(v, 0.f);
        h_e_out[ob + nt * 16] = v;
        if (do_scatter) {
          atomicAdd(&e_bar_i[(size_t)ii.x * D_FEAT + nt * 16 + lm], v);
          atomicAdd(&e_bar_i[(size_t)ii.y * D_FEAT + nt * 16 + lm], v);
          ebar_acc[nt] += v;
        }
      }
    }
  }

  if (do_scatter) {
#pragma unroll
    for (int nt = 0; nt < 4; nt++) atomicAdd(&sEbar[nt * 16 + lm], ebar_acc[nt]);
    __syncthreads();
    if (tid < D_FEAT) atomicAdd(&e_bar[tid], sEbar[tid]);
  }
}

// ---------------------------------------------------------------------------
// Vertex kernel: h_v = relu( [h_v | h_v0 | e_bar_i] @ Wcat + gvec ), + v_bar
// ---------------------------------------------------------------------------
__global__ void __launch_bounds__(256) vertex_kernel(
    float* __restrict__ h_v, const float* __restrict__ h_v0,
    const float* __restrict__ e_bar_i, const __bf16* __restrict__ wfrag,
    const float* __restrict__ gvec, float* __restrict__ v_bar, int n_atoms) {
  __shared__ __align__(16) __bf16 sW[WFRAG_TOTAL];
  __shared__ float sVbar[D_FEAT];

  int tid = threadIdx.x;
  {
    const uint4* src = (const uint4*)wfrag;
    uint4* dst = (uint4*)sW;
    for (int i = tid; i < (WFRAG_TOTAL * 2) / 16; i += 256) dst[i] = src[i];
  }
  if (tid < D_FEAT) sVbar[tid] = 0.f;
  __syncthreads();

  int lane = tid & 31;
  int wave = tid >> 5;
  int halfsel = lane >> 4;
  int lm = lane & 15;
  int ntiles = n_atoms / TILE_M;
  int wstride = gridDim.x * WAVES_PER_BLOCK;

  float gvals[4], vbar_acc[4] = {0.f, 0.f, 0.f, 0.f};
#pragma unroll
  for (int nt = 0; nt < 4; nt++) gvals[nt] = gvec[nt * 16 + lm];

  for (int tile = blockIdx.x * WAVES_PER_BLOCK + wave; tile < ntiles; tile += wstride) {
    int a0 = tile * TILE_M;
    int myrow = a0 + lm;

    v8f acc[4] = {};
    wmma_source_accum(sW, 0, h_v + (size_t)myrow * D_FEAT, halfsel, lane, acc);
    wmma_source_accum(sW, 1, h_v0 + (size_t)myrow * D_FEAT, halfsel, lane, acc);
    wmma_source_accum(sW, 2, e_bar_i + (size_t)myrow * D_FEAT, halfsel, lane, acc);

    int rbase = a0 + halfsel * 8;
#pragma unroll
    for (int r = 0; r < 8; r++) {
      size_t ob = (size_t)(rbase + r) * D_FEAT + lm;
#pragma unroll
      for (int nt = 0; nt < 4; nt++) {
        float v = fmaxf(acc[nt][r] + gvals[nt], 0.f);
        h_v[ob + nt * 16] = v;
        vbar_acc[nt] += v;
      }
    }
  }

#pragma unroll
  for (int nt = 0; nt < 4; nt++) atomicAdd(&sVbar[nt * 16 + lm], vbar_acc[nt]);
  __syncthreads();
  if (tid < D_FEAT) atomicAdd(&v_bar[tid], sVbar[tid]);
}

// ---------------------------------------------------------------------------
// Small prep / per-round kernels
// ---------------------------------------------------------------------------
__global__ void colsum_kernel(const float* __restrict__ atoms,
                              float* __restrict__ atom_sum, int total) {
  __shared__ float s[16];
  if (threadIdx.x < 16) s[threadIdx.x] = 0.f;
  __syncthreads();
  for (int i = blockIdx.x * blockDim.x + threadIdx.x; i < total;
       i += gridDim.x * blockDim.x)
    atomicAdd(&s[i & 15], atoms[i]);
  __syncthreads();
  if (threadIdx.x < 16) atomicAdd(&atom_sum[threadIdx.x], s[threadIdx.x]);
}

// h_u0 = atom_sum @ W_fu ; h_u = h_u0 ; c_e0 = W_fe@We1 ; c_e01 = W_fe@(We0+We1)
__global__ void prep_small_kernel(const float* __restrict__ atom_sum,
                                  const float* __restrict__ W_fu,
                                  const float* __restrict__ W_fe,
                                  const float* __restrict__ W_e,
                                  float* __restrict__ h_u, float* __restrict__ h_u0,
                                  float* __restrict__ c_e0, float* __restrict__ c_e01) {
  int n = threadIdx.x;
  float hu0 = 0.f;
  for (int k = 0; k < 16; k++) hu0 += atom_sum[k] * W_fu[k * 64 + n];
  h_u0[n] = hu0;
  h_u[n] = hu0;
  const float* We0 = W_e;
  const float* We1 = W_e + 4096;
  float c0 = 0.f, c01 = 0.f;
  for (int d = 0; d < 64; d++) {
    float f = W_fe[d];
    c0 += f * We1[d * 64 + n];
    c01 += f * (We0[d * 64 + n] + We1[d * 64 + n]);
  }
  c_e0[n] = c0;
  c_e01[n] = c01;
}

// h_v0 = atoms @ W_fv ; h_v = h_v0
__global__ void __launch_bounds__(256) init_hv_kernel(
    const float* __restrict__ atoms, const float* __restrict__ W_fv,
    float* __restrict__ h_v, float* __restrict__ h_v0, int n_atoms) {
  __shared__ float sW[16 * 64];
  for (int i = threadIdx.x; i < 1024; i += 256) sW[i] = W_fv[i];
  __syncthreads();
  int a = blockIdx.x * 4 + (threadIdx.x >> 6);
  int n = threadIdx.x & 63;
  if (a >= n_atoms) return;
  const float* ar = atoms + (size_t)a * 16;
  float acc = 0.f;
#pragma unroll
  for (int k = 0; k < 16; k++) acc += ar[k] * sW[k * 64 + n];
  h_v0[(size_t)a * 64 + n] = acc;
  h_v[(size_t)a * 64 + n] = acc;
}

// Pre-swizzle 3 edge mats (We0,We2,We3) and 3 vertex mats (Wv0,Wv1,Wv2) into
// the wave32 WMMA B-fragment layout [mat][kt][nt][lane][elem], f32 -> bf16.
// B element: k = kt*32 + (lane>=16 ? 8 : 0) + (elem<8 ? elem : elem+8) ; n = nt*16 + (lane&15)
__global__ void swizzle_kernel(const float* __restrict__ W_e,
                               const float* __restrict__ W_v,
                               __bf16* __restrict__ wfe, __bf16* __restrict__ wfv) {
  const int me[3] = {0, 2, 3};
  const int mv[3] = {0, 1, 2};
  for (int i = threadIdx.x; i < WFRAG_TOTAL; i += 256) {
    int elem = i & 15;
    int lane = (i >> 4) & 31;
    int nt = (i >> 9) & 3;
    int kt = (i >> 11) & 1;
    int mat = i >> 12;
    int k = kt * 32 + ((lane >= 16) ? 8 : 0) + ((elem < 8) ? elem : elem + 8);
    int n = nt * 16 + (lane & 15);
    wfe[i] = (__bf16)W_e[me[mat] * 4096 + k * 64 + n];
    wfv[i] = (__bf16)W_v[mv[mat] * 4096 + k * 64 + n];
  }
}

// g_e = h_u @ We4 + b_e ; g_v = h_u @ Wv3 + b_v   (uses pre-round h_u)
__global__ void gvec_kernel(const float* __restrict__ h_u,
                            const float* __restrict__ W_e, const float* __restrict__ W_v,
                            const float* __restrict__ b_e, const float* __restrict__ b_v,
                            float* __restrict__ g_e, float* __restrict__ g_v) {
  int n = threadIdx.x;
  const float* We4 = W_e + 4 * 4096;
  const float* Wv3 = W_v + 3 * 4096;
  float ge = b_e[n], gv = b_v[n];
  for (int k = 0; k < 64; k++) {
    float hu = h_u[k];
    ge += hu * We4[k * 64 + n];
    gv += hu * Wv3[k * 64 + n];
  }
  g_e[n] = ge;
  g_v[n] = gv;
}

// h_u = relu(h_u@Wu0 + h_u0@Wu1 + e_bar@Wu2 + v_bar@Wu3 + b_u)
__global__ void u_kernel(float* __restrict__ h_u, const float* __restrict__ h_u0,
                         const float* __restrict__ e_bar, const float* __restrict__ v_bar,
                         const float* __restrict__ W_u, const float* __restrict__ b_u) {
  __shared__ float s[64];
  int n = threadIdx.x;
  s[n] = h_u[n];
  __syncthreads();
  const float* Wu0 = W_u;
  const float* Wu1 = W_u + 4096;
  const float* Wu2 = W_u + 8192;
  const float* Wu3 = W_u + 12288;
  float acc = b_u[n];
  for (int k = 0; k < 64; k++)
    acc += s[k] * Wu0[k * 64 + n] + h_u0[k] * Wu1[k * 64 + n] +
           e_bar[k] * Wu2[k * 64 + n] + v_bar[k] * Wu3[k * 64 + n];
  h_u[n] = fmaxf(acc, 0.f);
}

// ---------------------------------------------------------------------------
extern "C" void kernel_launch(void* const* d_in, const int* in_sizes, int n_in,
                              void* d_out, int out_size, void* d_ws, size_t ws_size,
                              hipStream_t stream) {
  const float* atoms       = (const float*)d_in[0];
  const float* bond_orders = (const float*)d_in[1];
  const float* W_fe        = (const float*)d_in[2];
  const float* W_fv        = (const float*)d_in[3];
  const float* W_fu        = (const float*)d_in[4];
  const float* W_e         = (const float*)d_in[5];
  const float* b_e         = (const float*)d_in[6];
  const float* W_v         = (const float*)d_in[7];
  const float* b_v         = (const float*)d_in[8];
  const float* W_u         = (const float*)d_in[9];
  const float* b_u         = (const float*)d_in[10];
  const int*   bond_idxs   = (const int*)d_in[11];

  const int n_atoms = in_sizes[0] / 16;  // D_IN = 16
  const int n_bonds = in_sizes[1];

  float* h_e = (float*)d_out;  // h_e working buffer == output buffer (in-place rounds)
  float* ws = (float*)d_ws;

  float* h_v      = ws;
  float* h_v0     = ws + (size_t)n_atoms * 64;
  float* e_bar_i  = ws + (size_t)2 * n_atoms * 64;
  float* smallb   = ws + (size_t)3 * n_atoms * 64;
  float* h_u      = smallb + 0;
  float* h_u0     = smallb + 64;
  float* c_e0     = smallb + 128;
  float* c_e01    = smallb + 192;
  float* g_e      = smallb + 256;
  float* g_v      = smallb + 320;
  float* e_bar    = smallb + 384;
  float* v_bar    = smallb + 448;   // contiguous with e_bar (one memset)
  float* atom_sum = smallb + 512;
  __bf16* wfrag_e = (__bf16*)(smallb + 768);
  __bf16* wfrag_v = wfrag_e + WFRAG_TOTAL;

  // ---- prep ----
  hipMemsetAsync(atom_sum, 0, 16 * sizeof(float), stream);
  colsum_kernel<<<1024, 256, 0, stream>>>(atoms, atom_sum, n_atoms * 16);
  prep_small_kernel<<<1, 64, 0, stream>>>(atom_sum, W_fu, W_fe, W_e,
                                          h_u, h_u0, c_e0, c_e01);
  init_hv_kernel<<<(n_atoms + 3) / 4, 256, 0, stream>>>(atoms, W_fv, h_v, h_v0, n_atoms);
  swizzle_kernel<<<1, 256, 0, stream>>>(W_e, W_v, wfrag_e, wfrag_v);

  // ---- 3 message-passing rounds ----
  const int edge_blocks = 1568;   // ~4 bond tiles per wave
  const int vert_blocks = 392;
  for (int it = 0; it < 3; it++) {
    const int last = (it == 2);
    gvec_kernel<<<1, 64, 0, stream>>>(h_u, W_e, W_v, b_e, b_v, g_e, g_v);
    if (!last) {
      hipMemsetAsync(e_bar_i, 0, (size_t)n_atoms * 64 * sizeof(float), stream);
      hipMemsetAsync(e_bar, 0, 128 * sizeof(float), stream);  // e_bar + v_bar
    }
    edge_kernel<<<edge_blocks, 256, 0, stream>>>(
        h_e, h_e, h_v, bond_idxs, bond_orders, wfrag_e,
        (it == 0) ? c_e01 : c_e0, g_e, e_bar_i, e_bar,
        n_bonds, (it > 0) ? 1 : 0, last ? 0 : 1);
    if (!last) {
      vertex_kernel<<<vert_blocks, 256, 0, stream>>>(h_v, h_v0, e_bar_i, wfrag_v,
                                                     g_v, v_bar, n_atoms);
      u_kernel<<<1, 64, 0, stream>>>(h_u, h_u0, e_bar, v_bar, W_u, b_u);
    }
  }
}
